// SelfAttention_81501299409204
// MI455X (gfx1250) — matrix operand order repaired and checked by reading, not compiled, not run on hardware
//
#include <hip/hip_runtime.h>
#include <hip/hip_bf16.h>

typedef __attribute__((ext_vector_type(16))) _Float16 v16h;
typedef __attribute__((ext_vector_type(8)))  _Float16 v8h;
typedef __attribute__((ext_vector_type(8)))  float    v8f;

#define C_DIM 256
#define HW_DIM 4096
#define B_DIM 8
#define GROUPS 8

__device__ __forceinline__ v8f vzero8() {
    v8f z;
#pragma unroll
    for (int i = 0; i < 8; ++i) z[i] = 0.0f;
    return z;
}

__device__ __forceinline__ v8f wmma_f16(v16h a, v16h b, v8f c) {
    return __builtin_amdgcn_wmma_f32_16x16x32_f16(false, a, false, b, (short)0, c,
                                                  false, false);
}

// A-matrix 16x32 f16 fragment from row-major [row][k] (ld halves per row).
// Lane l: M = l&15; halves 0..15 -> K = (l>>4)*16 + h
__device__ __forceinline__ v16h load_a_frag(const _Float16* base, int ld, int lane) {
    int row = lane & 15, g = lane >> 4;
    const _Float16* p = base + (size_t)row * ld;
    v8h lo = *(const v8h*)(p + g * 16);
    v8h hi = *(const v8h*)(p + g * 16 + 8);
    v16h a;
#pragma unroll
    for (int i = 0; i < 8; ++i) { a[i] = lo[i]; a[i + 8] = hi[i]; }
    return a;
}

// ---------------- GroupNorm statistics ----------------
__global__ void __launch_bounds__(256) gn_stats_kernel(const float* __restrict__ x,
                                                       float* __restrict__ mean,
                                                       float* __restrict__ rstd) {
    __shared__ float ssum[256];
    __shared__ float ssq[256];
    int bg = blockIdx.x;                    // 0..63 : (b*8+g), groups contiguous in C
    int tid = threadIdx.x;
    const size_t CNT = (size_t)(C_DIM / GROUPS) * HW_DIM;  // 131072
    const float* p = x + (size_t)bg * CNT;
    float s = 0.f, sq = 0.f;
    for (size_t i = tid; i < CNT; i += 256) {
        float v = p[i];
        s += v; sq += v * v;
    }
    ssum[tid] = s; ssq[tid] = sq;
    __syncthreads();
    for (int st = 128; st > 0; st >>= 1) {
        if (tid < st) { ssum[tid] += ssum[tid + st]; ssq[tid] += ssq[tid + st]; }
        __syncthreads();
    }
    if (tid == 0) {
        float mu = ssum[0] / (float)CNT;
        float var = ssq[0] / (float)CNT - mu * mu;
        mean[bg] = mu;
        rstd[bg] = rsqrtf(var + 1e-5f);
    }
}

// ---------------- Normalize -> token-major f16 ----------------
__global__ void __launch_bounds__(256) gn_apply_kernel(const float* __restrict__ x,
                                                       const float* __restrict__ nw,
                                                       const float* __restrict__ nb,
                                                       const float* __restrict__ mean,
                                                       const float* __restrict__ rstd,
                                                       _Float16* __restrict__ xn_t) {
    size_t idx = (size_t)blockIdx.x * 256 + threadIdx.x;  // < 8*256*4096 = 2^23
    int n = (int)(idx & 4095);
    int c = (int)((idx >> 12) & 255);
    int b = (int)(idx >> 20);
    int bg = b * GROUPS + (c >> 5);
    float v = (x[idx] - mean[bg]) * rstd[bg] * nw[c] + nb[c];
    xn_t[((size_t)b * HW_DIM + n) * C_DIM + c] = (_Float16)v;
}

// ---------------- Weight conversion to f16 ----------------
__global__ void __launch_bounds__(256) cvt_w_kernel(const float* __restrict__ qkvw,
                                                    const float* __restrict__ projw,
                                                    _Float16* __restrict__ qkvw16,
                                                    _Float16* __restrict__ projw16) {
    int i = blockIdx.x * 256 + threadIdx.x;
    if (i < 3 * C_DIM * C_DIM) qkvw16[i] = (_Float16)qkvw[i];
    int j = i - 3 * C_DIM * C_DIM;
    if (j >= 0 && j < C_DIM * C_DIM) projw16[j] = (_Float16)projw[j];
}

// ---------------- QKV GEMM: 4 waves/block share weight tiles via LDS ----------------
// Block = 128 threads = 4 waves, 64 tokens. Per 16-output tile:
//   barrier -> cooperative stage W[16][256] (8KB) -> barrier -> 8 WMMAs from LDS.
// q,k stored token-major [b][n][c]; v stored channel-major [b][c][n]. q pre-scaled by 1/16.
__global__ void __launch_bounds__(128) qkv_gemm_kernel(const _Float16* __restrict__ xn_t,
                                                       const _Float16* __restrict__ w16,
                                                       const float* __restrict__ bias,
                                                       _Float16* __restrict__ q_tok,
                                                       _Float16* __restrict__ k_tok,
                                                       _Float16* __restrict__ vT) {
    __shared__ _Float16 wtile[16][C_DIM];   // 8 KB
    int tid = threadIdx.x;
    int lane = tid & 31, w = tid >> 5;
    int blk = blockIdx.x;                    // 0..511
    int b = blk >> 6;
    int t0 = ((blk & 63) << 6) + w * 16;     // this wave's 16 tokens
    const _Float16* xb = xn_t + ((size_t)b * HW_DIM + t0) * C_DIM;

    v16h af[8];
#pragma unroll
    for (int cc = 0; cc < 8; ++cc) af[cc] = load_a_frag(xb + cc * 32, C_DIM, lane);

    int n = lane & 15, g = lane >> 4;
    for (int ot = 0; ot < 48; ++ot) {
        __syncthreads();
        {
            const uint4* gw = (const uint4*)(w16 + (size_t)(ot * 16) * C_DIM);
            uint4* sw = (uint4*)&wtile[0][0];
#pragma unroll
            for (int j = 0; j < 4; ++j) sw[tid + 128 * j] = gw[tid + 128 * j];
        }
        __syncthreads();
        if (ot + 1 < 48)
            __builtin_prefetch(w16 + (size_t)((ot + 1) * 16) * C_DIM + tid * 64, 0, 1);

        v8f acc = vzero8();
#pragma unroll
        for (int cc = 0; cc < 8; ++cc) {
            v16h bf = *(const v16h*)(&wtile[n][0] + cc * 32 + g * 16);
            acc = wmma_f16(af[cc], bf, acc);
        }
        int o = ot * 16 + n;
        float bv = bias[o];
#pragma unroll
        for (int r = 0; r < 8; ++r) {
            int token = t0 + r + 8 * g;
            float v = acc[r] + bv;
            if (o < C_DIM) {
                q_tok[((size_t)b * HW_DIM + token) * C_DIM + o] = (_Float16)(v * 0.0625f);
            } else if (o < 2 * C_DIM) {
                k_tok[((size_t)b * HW_DIM + token) * C_DIM + (o - C_DIM)] = (_Float16)v;
            } else {
                vT[((size_t)b * C_DIM + (o - 2 * C_DIM)) * HW_DIM + token] = (_Float16)v;
            }
        }
    }
}

// ---------------- Flash attention: 4 waves/block share K/V tiles in LDS ----------------
__global__ void __launch_bounds__(128) attn_flash_kernel(const _Float16* __restrict__ q_tok,
                                                         const _Float16* __restrict__ k_tok,
                                                         const _Float16* __restrict__ vT,
                                                         _Float16* __restrict__ attn_out) {
    __shared__ _Float16 kbuf[32][C_DIM];      // 16 KB  [key][chan]
    __shared__ _Float16 vbuf[C_DIM][32];      // 16 KB  [chan][key]
    __shared__ _Float16 p_lds[4][16][32];     //  4 KB  per-wave P staging

    int tid = threadIdx.x;
    int lane = tid & 31;
    int w = tid >> 5;                          // wave id 0..3
    int blk = blockIdx.x;                      // 0..511
    int b = blk >> 6;                          // 64 query-blocks per batch
    int q0 = ((blk & 63) << 6) + w * 16;       // this wave's 16 queries
    const _Float16* qb = q_tok + (size_t)b * HW_DIM * C_DIM;
    const _Float16* kb = k_tok + (size_t)b * HW_DIM * C_DIM;
    const _Float16* vb = vT + (size_t)b * C_DIM * HW_DIM;
    int n = lane & 15, g = lane >> 4;

    // Q fragments: 16 queries x 256 channels (8 chunks of K=32), kept in VGPRs
    v16h qf[8];
#pragma unroll
    for (int cc = 0; cc < 8; ++cc)
        qf[cc] = load_a_frag(qb + (size_t)q0 * C_DIM + cc * 32, C_DIM, lane);

    v8f acc[16];
#pragma unroll
    for (int t = 0; t < 16; ++t) acc[t] = vzero8();
    float m[8], lsum[8];
#pragma unroll
    for (int r = 0; r < 8; ++r) { m[r] = -__builtin_inff(); lsum[r] = 0.f; }

    for (int k0 = 0; k0 < HW_DIM; k0 += 32) {
        // ---- cooperative stage: K tile is one contiguous 16KB block ----
        __syncthreads();   // previous chunk's compute done before overwrite
        {
            const uint4* gk = (const uint4*)(kb + (size_t)k0 * C_DIM);
            uint4* sk = (uint4*)&kbuf[0][0];
#pragma unroll
            for (int j = 0; j < 4; ++j) sk[tid + 128 * j] = gk[tid + 128 * j];
            // V: 256 channel rows of 64B (stride HW in global)
#pragma unroll
            for (int rr = 0; rr < 2; ++rr) {
                int c = tid * 2 + rr;
                const uint4* gv = (const uint4*)(vb + (size_t)c * HW_DIM + k0);
                uint4* sv = (uint4*)&vbuf[c][0];
#pragma unroll
                for (int j = 0; j < 4; ++j) sv[j] = gv[j];
            }
        }
        __syncthreads();   // tiles visible to all waves

        // prefetch next chunk while we compute this one (global_prefetch_b8)
        if (k0 + 32 < HW_DIM) {
            __builtin_prefetch(kb + (size_t)(k0 + 32) * C_DIM + tid * 64, 0, 1);
            __builtin_prefetch(vb + (size_t)(tid * 2) * HW_DIM + k0 + 32, 0, 1);
            __builtin_prefetch(vb + (size_t)(tid * 2 + 1) * HW_DIM + k0 + 32, 0, 1);
        }

        // ---- scores: S[16q x 32k] from LDS-resident K ----
        v8f s0 = vzero8(), s1 = vzero8();
#pragma unroll
        for (int cc = 0; cc < 8; ++cc) {
            v16h b0 = *(const v16h*)(&kbuf[n][0] + cc * 32 + g * 16);
            v16h b1 = *(const v16h*)(&kbuf[16 + n][0] + cc * 32 + g * 16);
            s0 = wmma_f16(qf[cc], b0, s0);
            s1 = wmma_f16(qf[cc], b1, s1);
        }

        // ---- online softmax (rows M = r + 8*g, spread across 16 lanes in N) ----
        float mnew[8], alpha[8];
#pragma unroll
        for (int r = 0; r < 8; ++r) {
            float mx = fmaxf(s0[r], s1[r]);
            mx = fmaxf(mx, __shfl_xor(mx, 1));
            mx = fmaxf(mx, __shfl_xor(mx, 2));
            mx = fmaxf(mx, __shfl_xor(mx, 4));
            mx = fmaxf(mx, __shfl_xor(mx, 8));
            mnew[r] = fmaxf(m[r], mx);
            alpha[r] = __expf(m[r] - mnew[r]);
        }
        float p0[8], p1[8];
#pragma unroll
        for (int r = 0; r < 8; ++r) {
            p0[r] = __expf(s0[r] - mnew[r]);
            p1[r] = __expf(s1[r] - mnew[r]);
            float rs = p0[r] + p1[r];
            rs += __shfl_xor(rs, 1);
            rs += __shfl_xor(rs, 2);
            rs += __shfl_xor(rs, 4);
            rs += __shfl_xor(rs, 8);
            lsum[r] = lsum[r] * alpha[r] + rs;
            m[r] = mnew[r];
        }
#pragma unroll
        for (int t = 0; t < 16; ++t)
#pragma unroll
            for (int r = 0; r < 8; ++r) acc[t][r] *= alpha[r];

        // ---- stage P (16x32 f16) through per-wave LDS to re-fragment as A ----
#pragma unroll
        for (int r = 0; r < 8; ++r) {
            p_lds[w][r + 8 * g][n] = (_Float16)p0[r];
            p_lds[w][r + 8 * g][16 + n] = (_Float16)p1[r];
        }
        asm volatile("s_wait_dscnt 0x0" ::: "memory");  // same-wave DS is in-order; be explicit
        v16h pf;
        {
            const _Float16* pr = &p_lds[w][n][0];
            v8h lo = *(const v8h*)(pr + g * 16);
            v8h hi = *(const v8h*)(pr + g * 16 + 8);
#pragma unroll
            for (int i = 0; i < 8; ++i) { pf[i] = lo[i]; pf[i + 8] = hi[i]; }
        }
        // ---- O(16x256) += P(16x32) x V(32x16 per channel tile), V from LDS ----
#pragma unroll
        for (int t = 0; t < 16; ++t) {
            v16h bv = *(const v16h*)(&vbuf[t * 16 + n][0] + g * 16);
            acc[t] = wmma_f16(pf, bv, acc[t]);
        }
    }

    // epilogue: divide by row sums, write token-major f16
#pragma unroll
    for (int r = 0; r < 8; ++r) {
        float rl = 1.0f / lsum[r];
        int token = q0 + r + 8 * g;
#pragma unroll
        for (int t = 0; t < 16; ++t) {
            attn_out[((size_t)b * HW_DIM + token) * C_DIM + t * 16 + n] =
                (_Float16)(acc[t][r] * rl);
        }
    }
}

// ---------------- Proj GEMM + bias + residual: 4 waves/block, LDS weight tiles ----------------
__global__ void __launch_bounds__(128) proj_gemm_kernel(const _Float16* __restrict__ attn_out,
                                                        const _Float16* __restrict__ pw16,
                                                        const float* __restrict__ pbias,
                                                        const float* __restrict__ x,
                                                        float* __restrict__ out) {
    __shared__ _Float16 wtile[16][C_DIM];   // 8 KB
    int tid = threadIdx.x;
    int lane = tid & 31, w = tid >> 5;
    int blk = blockIdx.x;                    // 0..511
    int b = blk >> 6;
    int t0 = ((blk & 63) << 6) + w * 16;
    const _Float16* ab = attn_out + ((size_t)b * HW_DIM + t0) * C_DIM;

    v16h af[8];
#pragma unroll
    for (int cc = 0; cc < 8; ++cc) af[cc] = load_a_frag(ab + cc * 32, C_DIM, lane);

    int n = lane & 15, g = lane >> 4;
    for (int ot = 0; ot < 16; ++ot) {
        __syncthreads();
        {
            const uint4* gw = (const uint4*)(pw16 + (size_t)(ot * 16) * C_DIM);
            uint4* sw = (uint4*)&wtile[0][0];
#pragma unroll
            for (int j = 0; j < 4; ++j) sw[tid + 128 * j] = gw[tid + 128 * j];
        }
        __syncthreads();
        if (ot + 1 < 16)
            __builtin_prefetch(pw16 + (size_t)((ot + 1) * 16) * C_DIM + tid * 64, 0, 1);

        v8f acc = vzero8();
#pragma unroll
        for (int cc = 0; cc < 8; ++cc) {
            v16h bf = *(const v16h*)(&wtile[n][0] + cc * 32 + g * 16);
            acc = wmma_f16(af[cc], bf, acc);
        }
        int o = ot * 16 + n;
        float bv = pbias[o];
#pragma unroll
        for (int r = 0; r < 8; ++r) {
            int token = t0 + r + 8 * g;
            size_t idx = ((size_t)b * C_DIM + o) * HW_DIM + token;
            out[idx] = x[idx] + acc[r] + bv;
        }
    }
}

extern "C" void kernel_launch(void* const* d_in, const int* in_sizes, int n_in,
                              void* d_out, int out_size, void* d_ws, size_t ws_size,
                              hipStream_t stream) {
    const float* x      = (const float*)d_in[0];
    const float* norm_w = (const float*)d_in[1];
    const float* norm_b = (const float*)d_in[2];
    const float* qkv_w  = (const float*)d_in[3];
    const float* qkv_b  = (const float*)d_in[4];
    const float* proj_w = (const float*)d_in[5];
    const float* proj_b = (const float*)d_in[6];
    float* out = (float*)d_out;

    char* ws = (char*)d_ws;
    const size_t HALF_PLANE = (size_t)B_DIM * HW_DIM * C_DIM * sizeof(_Float16); // 16 MB
    _Float16* xn_t     = (_Float16*)(ws);
    _Float16* q_tok    = (_Float16*)(ws + 1 * HALF_PLANE);
    _Float16* k_tok    = (_Float16*)(ws + 2 * HALF_PLANE);
    _Float16* vT       = (_Float16*)(ws + 3 * HALF_PLANE);
    _Float16* attn_out = (_Float16*)(ws + 4 * HALF_PLANE);
    _Float16* qkvw16   = (_Float16*)(ws + 5 * HALF_PLANE);
    _Float16* projw16  = (_Float16*)(ws + 5 * HALF_PLANE + (size_t)3 * C_DIM * C_DIM * sizeof(_Float16));
    float* mean        = (float*)(ws + 5 * HALF_PLANE + (size_t)4 * C_DIM * C_DIM * sizeof(_Float16));
    float* rstd        = mean + 64;

    gn_stats_kernel<<<B_DIM * GROUPS, 256, 0, stream>>>(x, mean, rstd);

    const size_t total = (size_t)B_DIM * C_DIM * HW_DIM;
    gn_apply_kernel<<<(int)(total / 256), 256, 0, stream>>>(x, norm_w, norm_b, mean, rstd, xn_t);

    cvt_w_kernel<<<(3 * C_DIM * C_DIM + C_DIM * C_DIM) / 256, 256, 0, stream>>>(
        qkv_w, proj_w, qkvw16, projw16);

    const int gblocks = B_DIM * (HW_DIM / 64);  // 512 blocks x 128 threads
    qkv_gemm_kernel<<<gblocks, 128, 0, stream>>>(xn_t, qkvw16, qkv_b, q_tok, k_tok, vT);

    attn_flash_kernel<<<gblocks, 128, 0, stream>>>(q_tok, k_tok, vT, attn_out);

    proj_gemm_kernel<<<gblocks, 128, 0, stream>>>(attn_out, projw16, proj_b, x, out);
}
